// HSTUMultiHeadAttention_83794811945339
// MI455X (gfx1250) — matrix operand order repaired and checked by reading, not compiled, not run on hardware
//
#include <hip/hip_runtime.h>
#include <math.h>

// ---------------- problem constants ----------------
#define B_    2
#define S_    2048
#define HID_  1024
#define NH_   8
#define LD_   64
#define AD_   64
#define D1_   2048            // uvqk output columns (U|V|Q|K)
#define M_    (B_*S_)         // 4096 tokens
#define NBUCK_ 128
#define EPS_  1e-5f

typedef __attribute__((ext_vector_type(16))) __bf16 v16bf;
typedef __attribute__((ext_vector_type(8)))  __bf16 v8bf;
typedef __attribute__((ext_vector_type(8)))  float  v8f;
typedef __attribute__((ext_vector_type(4)))  unsigned int u32x4;
typedef __attribute__((ext_vector_type(8)))  int i32x8;
typedef __attribute__((ext_vector_type(4)))  int i32x4;

#define AS3 __attribute__((address_space(3)))

#if __has_builtin(__builtin_amdgcn_global_load_async_to_lds_b128)
#define HAVE_ASYNC_LDS 1
#endif
#if __has_builtin(__builtin_amdgcn_tensor_load_to_lds)
#define HAVE_TDM 1
#endif

// ---------------- bf16 helpers ----------------
__device__ __forceinline__ unsigned short f2bfu(float f) {
  unsigned u = __builtin_bit_cast(unsigned, f);
  unsigned r = u + 0x7FFFu + ((u >> 16) & 1u);   // round-to-nearest-even
  return (unsigned short)(r >> 16);
}
__device__ __forceinline__ float bfu2f(unsigned short h) {
  unsigned u = ((unsigned)h) << 16;
  return __builtin_bit_cast(float, u);
}
__device__ __forceinline__ float fast_rcp(float x) { return __builtin_amdgcn_rcpf(x); }
__device__ __forceinline__ float siluf(float x) {
  return x * fast_rcp(1.f + __expf(-x));          // v_exp + v_rcp, no IEEE div chain
}

// ---------------- async / TDM staging helpers ----------------
__device__ __forceinline__ void copy16(const unsigned short* g, unsigned short* l) {
#ifdef HAVE_ASYNC_LDS
  __builtin_amdgcn_global_load_async_to_lds_b128((i32x4*)g, (i32x4*)l, 0, 0);
#else
  *(uint4*)l = *(const uint4*)g;
#endif
}
__device__ __forceinline__ void copy_wait() {
#ifdef HAVE_ASYNC_LDS
#if __has_builtin(__builtin_amdgcn_s_wait_asynccnt)
  __builtin_amdgcn_s_wait_asynccnt(0);
#else
  asm volatile("s_wait_asynccnt 0x0" ::: "memory");
#endif
#endif
}

#ifdef HAVE_TDM
// 2D tile load via Tensor Data Mover (D# per ISA 08 §8.3/8.4). One wave issues it.
__device__ __forceinline__ void tdm_load_2d(unsigned lds_off, const void* gaddr,
                                            unsigned tile_w, unsigned tile_h,
                                            unsigned row_stride) {
  unsigned long long ga = (unsigned long long)gaddr;
  u32x4 g0;
  g0[0] = 1u;                                             // count=1, user mode
  g0[1] = lds_off;                                        // lds_addr (bytes)
  g0[2] = (unsigned)ga;                                   // global_addr[31:0]
  g0[3] = (unsigned)((ga >> 32) & 0x01FFFFFFu) | (2u << 30);  // addr[56:32] | type=2
  i32x8 g1;
  g1[0] = 0x10000;                                        // data_size=1 (2 bytes)
  g1[1] = (int)((row_stride & 0xFFFFu) << 16);            // tensor_dim0 lo16
  g1[2] = (int)((row_stride >> 16) | ((M_ & 0xFFFFu) << 16)); // dim0 hi | dim1 lo
  g1[3] = (int)(((unsigned)M_ >> 16) | (tile_w << 16));   // dim1 hi | tile_dim0
  g1[4] = (int)tile_h;                                    // tile_dim1 (tile_dim2=0)
  g1[5] = (int)row_stride;                                // tensor_dim0_stride lo32
  g1[6] = 0;
  g1[7] = 0;
  i32x4 z = {0, 0, 0, 0};
#if defined(__clang_major__) && __clang_major__ >= 23
  i32x8 z8 = {0, 0, 0, 0, 0, 0, 0, 0};
  __builtin_amdgcn_tensor_load_to_lds(g0, g1, z, z, z8, 0);
#else
  __builtin_amdgcn_tensor_load_to_lds(g0, g1, z, z, 0);
#endif
}
#endif

// ---------------- WMMA fragment loaders (ISA 7.12.2, wave32) ----------------
__device__ __forceinline__ v16bf concat8(v8bf lo, v8bf hi) {
  return __builtin_shufflevector(lo, hi, 0, 1, 2, 3, 4, 5, 6, 7,
                                 8, 9, 10, 11, 12, 13, 14, 15);
}
// A-matrix 16x32: lane m=lane&15 reads two contiguous 8-elem runs (K 0-7/16-23, +8 for hi lanes)
__device__ __forceinline__ v16bf load_a_frag(const unsigned short* p, int row0, int k0, int ld) {
  int lane = threadIdx.x & 31;
  int m = lane & 15, half = lane >> 4;
  const unsigned short* rp = p + (row0 + m) * ld + k0 + half * 8;
  v8bf lo = __builtin_bit_cast(v8bf, *(const uint4*)(rp));
  v8bf hi = __builtin_bit_cast(v8bf, *(const uint4*)(rp + 16));
  return concat8(lo, hi);
}
// B-matrix 32x16 from N-major storage [n][k]: lane n=lane&15 reads two contiguous 8-elem runs (K 0-7/16-23, +8 for hi lanes)
__device__ __forceinline__ v16bf load_bt_frag(const unsigned short* p, int n0, int k0, int ld) {
  int lane = threadIdx.x & 31;
  int n = lane & 15, half = lane >> 4;
  const unsigned short* rp = p + (n0 + n) * ld + k0 + half * 8;
  v8bf lo = __builtin_bit_cast(v8bf, *(const uint4*)(rp));
  v8bf hi = __builtin_bit_cast(v8bf, *(const uint4*)(rp + 16));
  return concat8(lo, hi);
}
__device__ __forceinline__ v8f wmma_bf(v16bf a, v16bf b, v8f c) {
  return __builtin_amdgcn_wmma_f32_16x16x32_bf16(false, a, false, b, (short)0, c, false, false);
}

// ---------------- kernel 0: fp32 -> bf16 transpose conversion (KxN -> NxK) ----------
__global__ void k_cvt_t(const float* __restrict__ src, unsigned short* __restrict__ dst,
                        int K, int nlog2) {
  int i = blockIdx.x * blockDim.x + threadIdx.x;
  int N = 1 << nlog2;
  if (i < K * N) {
    int k = i >> nlog2, n = i & (N - 1);
    dst[(size_t)n * K + k] = f2bfu(src[i]);
  }
}

// ---------------- kernel 1: FiLM gate table (4 action ids x 2048) ----------------
__global__ void k_rbtab(const float* __restrict__ ae, const float* __restrict__ lw,
                        const float* __restrict__ lb, const float* __restrict__ fw,
                        const float* __restrict__ fb, float* __restrict__ rb) {
  int id = blockIdx.x;
  const float* a = ae + id * 32;
  float s = 0.f, ss = 0.f;
#pragma unroll
  for (int k = 0; k < 32; ++k) { float t = a[k]; s += t; ss += t * t; }
  float mean = s * (1.f / 32.f);
  float var  = ss * (1.f / 32.f) - mean * mean;
  float inv  = rsqrtf(var + EPS_);
  for (int n = threadIdx.x; n < 2 * HID_; n += blockDim.x) {
    float acc = fb[n];
#pragma unroll
    for (int k = 0; k < 32; ++k)
      acc += ((a[k] - mean) * inv * lw[k] + lb[k]) * fw[k * (2 * HID_) + n];
    rb[id * (2 * HID_) + n] = acc;
  }
}

// ---------------- kernel 2: input LayerNorm -> bf16 ----------------
__global__ __launch_bounds__(256) void k_ln_in(const float* __restrict__ x,
                                               const float* __restrict__ w,
                                               const float* __restrict__ b,
                                               unsigned short* __restrict__ out) {
  int tok = blockIdx.x, tid = threadIdx.x;
  const float* row = x + (size_t)tok * HID_;
  __shared__ float red[256];
  float v[4]; float s = 0.f, ss = 0.f;
#pragma unroll
  for (int e = 0; e < 4; ++e) { float t = row[tid + e * 256]; v[e] = t; s += t; ss += t * t; }
  red[tid] = s; __syncthreads();
  for (int o = 128; o > 0; o >>= 1) { if (tid < o) red[tid] += red[tid + o]; __syncthreads(); }
  float mean = red[0] * (1.f / HID_); __syncthreads();
  red[tid] = ss; __syncthreads();
  for (int o = 128; o > 0; o >>= 1) { if (tid < o) red[tid] += red[tid + o]; __syncthreads(); }
  float var = red[0] * (1.f / HID_) - mean * mean;
  float inv = rsqrtf(var + EPS_);
#pragma unroll
  for (int e = 0; e < 4; ++e) {
    int i = tid + e * 256;
    out[(size_t)tok * HID_ + i] = f2bfu((v[e] - mean) * inv * w[i] + b[i]);
  }
}

// ---------------- kernel 3: WMMA GEMM  mm = silu(norm @ uvqk)  (4096x2048,K=1024) ----
// Bt is uvqk transposed (N x K), so both operands use contiguous b128 LDS reads.
__global__ __launch_bounds__(256) void k_gemm_uvqk(const unsigned short* __restrict__ A,
                                                   const unsigned short* __restrict__ Bt,
                                                   unsigned short* __restrict__ Cm) {
  __shared__ __align__(16) unsigned short As[128 * 32];
  __shared__ __align__(16) unsigned short Bs[128 * 32];
  const int tid = threadIdx.x, lane = tid & 31, wave = tid >> 5;
  const int wm = wave >> 1, wn = wave & 1;
  const int row0 = blockIdx.y * 128, col0 = blockIdx.x * 128;
  v8f acc[2][4];
#pragma unroll
  for (int i = 0; i < 2; ++i)
#pragma unroll
    for (int j = 0; j < 4; ++j) acc[i][j] = {};
  for (int k0 = 0; k0 < HID_; k0 += 32) {
    __syncthreads();
#pragma unroll
    for (int part = 0; part < 2; ++part) {          // A tile 128x32, B tile 128x32
      int idx = part * 256 + tid, r = idx >> 2, c = (idx & 3) * 8;
      copy16(A  + (size_t)(row0 + r) * HID_ + k0 + c, &As[r * 32 + c]);
      copy16(Bt + (size_t)(col0 + r) * HID_ + k0 + c, &Bs[r * 32 + c]);
    }
    copy_wait();
    __syncthreads();
    v16bf af[2], bfv[4];
#pragma unroll
    for (int i = 0; i < 2; ++i) af[i] = load_a_frag(As, wm * 32 + i * 16, 0, 32);
#pragma unroll
    for (int j = 0; j < 4; ++j) bfv[j] = load_bt_frag(Bs, wn * 64 + j * 16, 0, 32);
#pragma unroll
    for (int i = 0; i < 2; ++i)
#pragma unroll
      for (int j = 0; j < 4; ++j) acc[i][j] = wmma_bf(af[i], bfv[j], acc[i][j]);
  }
  const int mh = (lane >> 4) * 8, nl = lane & 15;
#pragma unroll
  for (int i = 0; i < 2; ++i)
#pragma unroll
    for (int j = 0; j < 4; ++j)
#pragma unroll
      for (int r = 0; r < 8; ++r) {
        int m = row0 + wm * 32 + i * 16 + mh + r;
        int n = col0 + wn * 64 + j * 16 + nl;
        Cm[(size_t)m * D1_ + n] = f2bfu(siluf(acc[i][j][r]));
      }
}

// ---------------- kernel 4: RoPE in place on Q/K regions of mm ----------------
__global__ void k_rope(unsigned short* __restrict__ mm, const float* __restrict__ inv_freq) {
  int t = blockIdx.x * blockDim.x + threadIdx.x;
  if (t >= M_ * NH_) return;
  int tok = t >> 3, h = t & 7;
  int s = tok & (S_ - 1);
  size_t baseq = (size_t)tok * D1_ + 2 * LD_ * NH_ + h * AD_;
  size_t basek = baseq + AD_ * NH_;
#pragma unroll
  for (int p = 0; p < 16; ++p) {
    float ang = (float)s * inv_freq[p];
    float c = __cosf(ang), si = __sinf(ang);
    {
      float e = bfu2f(mm[baseq + 2 * p]), o = bfu2f(mm[baseq + 2 * p + 1]);
      mm[baseq + 2 * p]     = f2bfu(e * c - o * si);
      mm[baseq + 2 * p + 1] = f2bfu(o * c + e * si);
    }
    {
      float e = bfu2f(mm[basek + 2 * p]), o = bfu2f(mm[basek + 2 * p + 1]);
      mm[basek + 2 * p]     = f2bfu(e * c - o * si);
      mm[basek + 2 * p + 1] = f2bfu(o * c + e * si);
    }
  }
}

// ---------------- kernel 4b: transpose V into [b,h,d,s] for N-major fragments ------
__global__ void k_transv(const unsigned short* __restrict__ mm, unsigned short* __restrict__ vt) {
  int t = blockIdx.x * blockDim.x + threadIdx.x;     // 16*64*2048 threads
  int s = t & (S_ - 1);
  int d = (t >> 11) & 63;
  int bh = t >> 17;
  int b = bh >> 3, h = bh & 7;
  vt[t] = mm[((size_t)b * S_ + s) * D1_ + LD_ * NH_ + h * AD_ + d];
}

// ---------------- kernel 5: fused attention ----------------
__global__ __launch_bounds__(128) void k_attn(const unsigned short* __restrict__ mm,
                                              const unsigned short* __restrict__ vt,
                                              const int* __restrict__ interval,
                                              const float* __restrict__ ts_w,
                                              const float* __restrict__ pos_w,
                                              unsigned short* __restrict__ udot) {
  __shared__ __align__(16) unsigned short Qs[64 * 64];
  __shared__ __align__(16) unsigned short Ks[32 * 64];   // [key][d]  (= N-major for Q*K^T)
  __shared__ __align__(16) unsigned short Vts[64 * 32];  // [d][key]  (= N-major for P*V)
  __shared__ __align__(16) unsigned short Ps[4][16 * 32];
  __shared__ __align__(16) float Os[4][16 * 64];
  const int tid = threadIdx.x, lane = tid & 31, wave = tid >> 5;
  const int bh = blockIdx.y, b = bh >> 3, h = bh & 7;
  const int q0 = blockIdx.x * 64;
  const size_t rowbase = (size_t)b * S_;
  const int qcol = 2 * LD_ * NH_ + h * AD_;
  const int kcol = qcol + AD_ * NH_;
  const int mh = (lane >> 4) * 8, nl = lane & 15;

  // stage Q tile (64x64) — TDM if available, else async/sync copy
#ifdef HAVE_TDM
  if (tid < 32) {
    unsigned qs_off = (unsigned)(unsigned long long)(AS3 unsigned short*)Qs;
    tdm_load_2d(qs_off, mm + (rowbase + q0) * D1_ + qcol, 64, 64, D1_);
    __builtin_amdgcn_s_wait_tensorcnt(0);
  }
#else
#pragma unroll
  for (int part = 0; part < 4; ++part) {
    int idx = part * 128 + tid, r = idx >> 3, c = (idx & 7) * 8;
    copy16(mm + (rowbase + q0 + r) * D1_ + qcol + c, &Qs[r * 64 + c]);
  }
  copy_wait();
#endif
  __syncthreads();
  v16bf qa0 = load_a_frag(Qs, wave * 16, 0, 64);
  v16bf qa1 = load_a_frag(Qs, wave * 16, 32, 64);
  v8f oacc[4];
#pragma unroll
  for (int j = 0; j < 4; ++j) oacc[j] = {};

  // hoist query-side interval values (ext[i+1]) for this lane's 8 rows
  int e1v[8];
#pragma unroll
  for (int r = 0; r < 8; ++r) {
    int qi = q0 + wave * 16 + mh + r;
    e1v[r] = interval[rowbase + ((qi + 1 < S_) ? (qi + 1) : (S_ - 1))];
  }

  const int ntile = blockIdx.x * 2 + 2;
  for (int kt = 0; kt < ntile; ++kt) {
    const int key0 = kt * 32;
    __syncthreads();
#pragma unroll
    for (int part = 0; part < 2; ++part) {           // K tile 32x64
      int idx = part * 128 + tid, r = idx >> 3, c = (idx & 7) * 8;
      copy16(mm + (rowbase + key0 + r) * D1_ + kcol + c, &Ks[r * 64 + c]);
    }
#pragma unroll
    for (int part = 0; part < 2; ++part) {           // Vt tile 64x32
      int idx = part * 128 + tid, r = idx >> 2, c = (idx & 3) * 8;
      copy16(vt + ((size_t)bh * 64 + r) * S_ + key0 + c, &Vts[r * 32 + c]);
    }
    copy_wait();
    __syncthreads();
#pragma unroll
    for (int sub = 0; sub < 2; ++sub) {              // two 16-key subtiles -> P(16x32)
      v8f sacc = {};
      v16bf kb0 = load_bt_frag(Ks, sub * 16, 0, 64);
      v16bf kb1 = load_bt_frag(Ks, sub * 16, 32, 64);
      sacc = wmma_bf(qa0, kb0, sacc);
      sacc = wmma_bf(qa1, kb1, sacc);
      int kj = key0 + sub * 16 + nl;
      int e0 = interval[rowbase + kj];               // key-side interval, 1 load/lane
#pragma unroll
      for (int r = 0; r < 8; ++r) {
        int qi = q0 + wave * 16 + mh + r;
        float x = sacc[r];
        float dt = fmaxf(fabsf((float)(e1v[r] - e0)), 1.0f);
        int bucket = (int)(__logf(dt) * (1.0f / 0.301f));
        bucket = bucket < 0 ? 0 : (bucket > NBUCK_ ? NBUCK_ : bucket);
        x += ts_w[bucket] + pos_w[kj - qi + S_ - 1];
        x = siluf(x) * (1.0f / (float)S_);
        if (kj > qi) x = 0.f;                        // causal mask
        Ps[wave][(mh + r) * 32 + sub * 16 + nl] = f2bfu(x);
      }
    }
    __syncthreads();                                 // C-layout -> A-layout via LDS
    v16bf pa = load_a_frag(Ps[wave], 0, 0, 32);
#pragma unroll
    for (int j = 0; j < 4; ++j) {
      v16bf vb = load_bt_frag(Vts, j * 16, 0, 32);
      oacc[j] = wmma_bf(pa, vb, oacc[j]);
    }
  }
  // epilogue: ln_noaffine per row of 64, then u_dot = U * out
#pragma unroll
  for (int j = 0; j < 4; ++j)
#pragma unroll
    for (int r = 0; r < 8; ++r)
      Os[wave][(mh + r) * 64 + j * 16 + nl] = oacc[j][r];
  __syncthreads();
  if (lane < 16) {
    const float* o = Os[wave] + lane * 64;
    float s = 0.f, ss = 0.f;
#pragma unroll
    for (int d = 0; d < 64; ++d) { float t = o[d]; s += t; ss += t * t; }
    float mean = s * (1.f / 64.f);
    float var  = ss * (1.f / 64.f) - mean * mean;
    float inv  = rsqrtf(var + EPS_);
    int qi = q0 + wave * 16 + lane;
    size_t urow = (rowbase + qi) * D1_ + h * LD_;
    size_t drow = (rowbase + qi) * (size_t)(LD_ * NH_) + h * LD_;
    for (int d = 0; d < 64; ++d) {
      float nrm = (o[d] - mean) * inv;
      udot[drow + d] = f2bfu(bfu2f(mm[urow + d]) * nrm);
    }
  }
}

// ---------------- kernel 6: WMMA GEMM  out = input + udot @ o_w + o_b (4096x1024,K=512)
__global__ __launch_bounds__(256) void k_gemm_out(const unsigned short* __restrict__ A,
                                                  const unsigned short* __restrict__ Bt,
                                                  const float* __restrict__ inp,
                                                  const float* __restrict__ ob,
                                                  float* __restrict__ out) {
  __shared__ __align__(16) unsigned short As[128 * 32];
  __shared__ __align__(16) unsigned short Bs[128 * 32];
  const int tid = threadIdx.x, lane = tid & 31, wave = tid >> 5;
  const int wm = wave >> 1, wn = wave & 1;
  const int row0 = blockIdx.y * 128, col0 = blockIdx.x * 128;
  v8f acc[2][4];
#pragma unroll
  for (int i = 0; i < 2; ++i)
#pragma unroll
    for (int j = 0; j < 4; ++j) acc[i][j] = {};
  for (int k0 = 0; k0 < LD_ * NH_; k0 += 32) {
    __syncthreads();
#pragma unroll
    for (int part = 0; part < 2; ++part) {
      int idx = part * 256 + tid, r = idx >> 2, c = (idx & 3) * 8;
      copy16(A  + (size_t)(row0 + r) * (LD_ * NH_) + k0 + c, &As[r * 32 + c]);
      copy16(Bt + (size_t)(col0 + r) * (LD_ * NH_) + k0 + c, &Bs[r * 32 + c]);
    }
    copy_wait();
    __syncthreads();
    v16bf af[2], bfv[4];
#pragma unroll
    for (int i = 0; i < 2; ++i) af[i] = load_a_frag(As, wm * 32 + i * 16, 0, 32);
#pragma unroll
    for (int j = 0; j < 4; ++j) bfv[j] = load_bt_frag(Bs, wn * 64 + j * 16, 0, 32);
#pragma unroll
    for (int i = 0; i < 2; ++i)
#pragma unroll
      for (int j = 0; j < 4; ++j) acc[i][j] = wmma_bf(af[i], bfv[j], acc[i][j]);
  }
  const int mh = (lane >> 4) * 8, nl = lane & 15;
#pragma unroll
  for (int i = 0; i < 2; ++i)
#pragma unroll
    for (int j = 0; j < 4; ++j)
#pragma unroll
      for (int r = 0; r < 8; ++r) {
        int m = row0 + wm * 32 + i * 16 + mh + r;
        int n = col0 + wn * 64 + j * 16 + nl;
        out[(size_t)m * HID_ + n] = acc[i][j][r] + inp[(size_t)m * HID_ + n] + ob[n];
      }
}

// ---------------- kernel 7: FiLM gating in place ----------------
__global__ __launch_bounds__(256) void k_film(float* __restrict__ out,
                                              const int* __restrict__ nat,
                                              const int* __restrict__ nmask,
                                              const float* __restrict__ pw,
                                              const float* __restrict__ pb,
                                              const float* __restrict__ rb,
                                              const float* __restrict__ rs,
                                              const float* __restrict__ bs) {
  int tok = blockIdx.x, tid = threadIdx.x;
  float* row = out + (size_t)tok * HID_;
  __shared__ float red[256];
  float v[4]; float s = 0.f, ss = 0.f;
#pragma unroll
  for (int e = 0; e < 4; ++e) { float t = row[tid + e * 256]; v[e] = t; s += t; ss += t * t; }
  red[tid] = s; __syncthreads();
  for (int o = 128; o > 0; o >>= 1) { if (tid < o) red[tid] += red[tid + o]; __syncthreads(); }
  float mean = red[0] * (1.f / HID_); __syncthreads();
  red[tid] = ss; __syncthreads();
  for (int o = 128; o > 0; o >>= 1) { if (tid < o) red[tid] += red[tid + o]; __syncthreads(); }
  float var = red[0] * (1.f / HID_) - mean * mean;
  float inv = rsqrtf(var + EPS_);
  int id = (nmask[tok] == 1) ? (nat[tok] + 1) : 0;
  float rscale = rs[0], bscale = bs[0];
#pragma unroll
  for (int e = 0; e < 4; ++e) {
    int i = tid + e * 256;
    float nrm = (v[e] - mean) * inv * pw[i] + pb[i];
    float r  = rb[id * (2 * HID_) + i];
    float bg = rb[id * (2 * HID_) + HID_ + i];
    float e2 = __expf(2.f * r);
    float th = (e2 - 1.f) * fast_rcp(e2 + 1.f);     // fast tanh
    row[i] = v[e] + nrm * th * rscale + bg * bscale;
  }
}

// ---------------- host launcher ----------------
extern "C" void kernel_launch(void* const* d_in, const int* in_sizes, int n_in,
                              void* d_out, int out_size, void* d_ws, size_t ws_size,
                              hipStream_t stream) {
  const float* input    = (const float*)d_in[0];
  const int*   interval = (const int*)d_in[1];
  // d_in[2] attn_mask is a deterministic causal tril -> computed analytically
  const int*   nat      = (const int*)d_in[3];
  const int*   nmask    = (const int*)d_in[4];
  const float* ln_w     = (const float*)d_in[5];
  const float* ln_b     = (const float*)d_in[6];
  const float* pin_w    = (const float*)d_in[7];
  const float* pin_b    = (const float*)d_in[8];
  const float* uvqk     = (const float*)d_in[9];
  const float* o_w      = (const float*)d_in[10];
  const float* o_b      = (const float*)d_in[11];
  const float* ts_w     = (const float*)d_in[12];
  const float* pos_w    = (const float*)d_in[13];
  const float* aemb     = (const float*)d_in[14];
  const float* flw      = (const float*)d_in[15];
  const float* flb      = (const float*)d_in[16];
  const float* fw       = (const float*)d_in[17];
  const float* fb       = (const float*)d_in[18];
  const float* rscale   = (const float*)d_in[19];
  const float* bscale   = (const float*)d_in[20];
  const float* inv_freq = (const float*)d_in[21];

  // workspace layout (~37 MB + 32 KB)
  char* ws = (char*)d_ws;
  unsigned short* normb  = (unsigned short*)(ws);                  // 8 MB  (4096x1024 bf16)
  unsigned short* uvqkbt = (unsigned short*)(ws + (8u  << 20));    // 4 MB  (2048x1024 bf16, N-major)
  unsigned short* owbt   = (unsigned short*)(ws + (12u << 20));    // 1 MB  (1024x512  bf16, N-major)
  unsigned short* mmb    = (unsigned short*)(ws + (13u << 20));    // 16 MB (4096x2048 bf16)
  unsigned short* udotb  = (unsigned short*)(ws + (29u << 20));    // 4 MB  (4096x512  bf16)
  unsigned short* vtb    = (unsigned short*)(ws + (33u << 20));    // 4 MB  (16x64x2048 bf16)
  float*          rbtab  = (float*)(ws + (37u << 20));             // 32 KB (4x2048 f32)

  float* out = (float*)d_out;

  k_cvt_t<<<(HID_ * D1_ + 255) / 256, 256, 0, stream>>>(uvqk, uvqkbt, HID_, 11);
  k_cvt_t<<<(LD_ * NH_ * HID_ + 255) / 256, 256, 0, stream>>>(o_w, owbt, LD_ * NH_, 10);
  k_rbtab<<<4, 256, 0, stream>>>(aemb, flw, flb, fw, fb, rbtab);
  k_ln_in<<<M_, 256, 0, stream>>>(input, ln_w, ln_b, normb);
  k_gemm_uvqk<<<dim3(D1_ / 128, M_ / 128), 256, 0, stream>>>(normb, uvqkbt, mmb);
  k_rope<<<(M_ * NH_) / 256, 256, 0, stream>>>(mmb, inv_freq);
  k_transv<<<(B_ * NH_ * LD_ * S_) / 256, 256, 0, stream>>>(mmb, vtb);
  k_attn<<<dim3(S_ / 64, B_ * NH_), 128, 0, stream>>>(mmb, vtb, interval, ts_w, pos_w, udotb);
  k_gemm_out<<<dim3(HID_ / 128, M_ / 128), 256, 0, stream>>>(udotb, owbt, input, o_b, out);
  k_film<<<M_, 256, 0, stream>>>(out, nat, nmask, pin_w, pin_b, rbtab, rscale, bscale);
}
